// AttentionBlock_4114578670332
// MI455X (gfx1250) — matrix operand order repaired and checked by reading, not compiled, not run on hardware
//
#include <hip/hip_runtime.h>
#include <hip/hip_bf16.h>

typedef unsigned short u16;
typedef __attribute__((ext_vector_type(16))) __bf16 v16bf;
typedef __attribute__((ext_vector_type(8)))  float  v8f;
typedef __attribute__((ext_vector_type(4)))  unsigned int u32x4;
typedef __attribute__((ext_vector_type(8)))  unsigned int u32x8;

union Frag  { uint4 q[2]; v16bf bf; };
union Pack8 { u16 u[8];  uint4 q;    };
union Pack16{ u16 u[16]; uint4 q[2]; };

__device__ __forceinline__ u16 f2bf(float f) {
  unsigned u = __float_as_uint(f);
  u += 0x7FFFu + ((u >> 16) & 1u);
  return (u16)(u >> 16);
}

#define WMMA_BF16(a, b, c) \
  __builtin_amdgcn_wmma_f32_16x16x32_bf16(false, (a), false, (b), (short)0, (c), false, false)

#define CCH   512
#define HW    4096
#define HEADS 8
#define HD    64

#define LDSOFF(p) ((unsigned)(unsigned long long)(p))

// ---- TDM: async 2-D bf16 tile load Global->LDS (CDNA5 ISA D# layout) ----
// tile = tile1 rows x tile0 elements, source row stride = stride0 elements.
// All tiles here have 64-element rows; pad codes (4,3) give LDS row stride
// 64+8=72 halfwords (144 B = 36 banks -> conflict-free b128 reads).
__device__ __forceinline__ void tdm_load_2d(unsigned lds_off, const u16* gptr,
                                            unsigned td0, unsigned td1,
                                            unsigned stride0,
                                            unsigned tile0, unsigned tile1) {
  unsigned long long ga = (unsigned long long)gptr;
  u32x4 g0;
  g0.x = 1u;                                            // count=1, user desc
  g0.y = lds_off;                                       // lds_addr [63:32]
  g0.z = (unsigned)ga;                                  // global_addr lo
  g0.w = ((unsigned)(ga >> 32) & 0x01FFFFFFu) | (2u << 30);  // addr hi | type=2
  u32x8 g1;
  g1[0] = (1u << 16) | (1u << 20) | (4u << 22) | (3u << 25); // 2B elems, pad_en, int=32DW, amt=4DW
  g1[1] = (td0 & 0xFFFFu) << 16;                        // tensor_dim0 [79:48]
  g1[2] = (td0 >> 16) | ((td1 & 0xFFFFu) << 16);        // tensor_dim1 [111:80]
  g1[3] = (td1 >> 16) | ((tile0 & 0xFFFFu) << 16);      // tile_dim0 [127:112]
  g1[4] = tile1 & 0xFFFFu;                              // tile_dim1; tile_dim2=0
  g1[5] = stride0;                                      // tensor_dim0_stride lo
  g1[6] = 0u;
  g1[7] = 0u;
  u32x4 gz; gz.x = 0u; gz.y = 0u; gz.z = 0u; gz.w = 0u;
  asm volatile("tensor_load_to_lds %0, %1, %2, %3"
               :: "s"(g0), "s"(g1), "s"(gz), "s"(gz) : "memory");
}

// ---------------- Kernel 1: GroupNorm -> hT[4096][512] bf16 -----------------
__global__ __launch_bounds__(256) void gn_kernel(const float* __restrict__ x,
                                                 const float* __restrict__ gw,
                                                 const float* __restrict__ gb,
                                                 u16* __restrict__ hT) {
  __shared__ float red[256];
  const int g = blockIdx.x, t = threadIdx.x;
  const float* xg = x + g * 16 * HW;
  float s = 0.f, s2 = 0.f;
  for (int i = t; i < 16 * HW; i += 256) { float v = xg[i]; s += v; s2 += v * v; }
  red[t] = s; __syncthreads();
  for (int o = 128; o > 0; o >>= 1) { if (t < o) red[t] += red[t + o]; __syncthreads(); }
  const float mean = red[0] * (1.0f / 65536.0f);
  __syncthreads();
  red[t] = s2; __syncthreads();
  for (int o = 128; o > 0; o >>= 1) { if (t < o) red[t] += red[t + o]; __syncthreads(); }
  const float var  = red[0] * (1.0f / 65536.0f) - mean * mean;
  const float rstd = rsqrtf(var + 1e-6f);
  float wv[16], bv[16];
  #pragma unroll
  for (int c = 0; c < 16; ++c) {
    wv[c] = gw[g * 16 + c] * rstd;
    bv[c] = gb[g * 16 + c] - mean * wv[c];
  }
  for (int p = t; p < HW; p += 256) {
    Pack16 pk;
    #pragma unroll
    for (int c = 0; c < 16; ++c)
      pk.u[c] = f2bf(xg[c * HW + p] * wv[c] + bv[c]);
    uint4* dst = (uint4*)(hT + p * CCH + g * 16);
    dst[0] = pk.q[0]; dst[1] = pk.q[1];
  }
}

// ---------------- Kernel 2: weights fp32 -> bf16 ----------------------------
__global__ __launch_bounds__(256) void cvt_w_kernel(const float* __restrict__ qkvw,
                                                    const float* __restrict__ projw,
                                                    u16* __restrict__ qw,
                                                    u16* __restrict__ pw) {
  int i = blockIdx.x * 256 + threadIdx.x;
  const int NQ = 3 * CCH * CCH;
  if (i < NQ) qw[i] = f2bf(qkvw[i]);
  else        pw[i - NQ] = f2bf(projw[i - NQ]);
}

// ---- shared GEMM core: C[128x64] tile, K=512, TDM-staged B, A双-buffered ----
template <typename EpilogueFn>
__device__ __forceinline__ void gemm_core(const u16* __restrict__ arow,
                                          const u16* __restrict__ bbase, // row nb of B^T [N][K]
                                          u16 (&Bs)[2][64 * 72],
                                          int w, int l16, int half,
                                          v8f (&acc)[4]) {
  Frag aF[2][2];
  aF[0][0].q[0] = *(const uint4*)(arow + half * 8);
  aF[0][0].q[1] = *(const uint4*)(arow + half * 8 + 16);
  aF[0][1].q[0] = *(const uint4*)(arow + 32 + half * 8);
  aF[0][1].q[1] = *(const uint4*)(arow + 32 + half * 8 + 16);
  if (w == 0)
    tdm_load_2d(LDSOFF(&Bs[0][0]), bbase, CCH, HW, CCH, 64, 64);
  #pragma unroll
  for (int ph = 0; ph < 8; ++ph) {
    const int kk = ph * 64;
    const int cur = ph & 1;
    if (ph + 1 < 8) {   // prefetch next A fragments (waited next iteration)
      const u16* an = arow + kk + 64;
      aF[cur ^ 1][0].q[0] = *(const uint4*)(an + half * 8);
      aF[cur ^ 1][0].q[1] = *(const uint4*)(an + half * 8 + 16);
      aF[cur ^ 1][1].q[0] = *(const uint4*)(an + 32 + half * 8);
      aF[cur ^ 1][1].q[1] = *(const uint4*)(an + 32 + half * 8 + 16);
    }
    __syncthreads();                       // Bs[cur^1] free for overwrite
    if (w == 0) {
      if (ph + 1 < 8) {
        tdm_load_2d(LDSOFF(&Bs[cur ^ 1][0]), bbase + kk + 64, CCH, HW, CCH, 64, 64);
        __builtin_amdgcn_s_wait_tensorcnt(1);   // current tile landed
      } else {
        __builtin_amdgcn_s_wait_tensorcnt(0);
      }
    }
    __syncthreads();                       // Bs[cur] visible to all waves
    // preload all 8 B fragments, then issue 8 WMMAs (partial dscnt waits)
    Frag b[4][2];
    #pragma unroll
    for (int nt = 0; nt < 4; ++nt) {
      const u16* kr = &Bs[cur][(nt * 16 + l16) * 72 + half * 8];
      b[nt][0].q[0] = *(const uint4*)(kr);      b[nt][0].q[1] = *(const uint4*)(kr + 16);
      b[nt][1].q[0] = *(const uint4*)(kr + 32); b[nt][1].q[1] = *(const uint4*)(kr + 48);
    }
    #pragma unroll
    for (int nt = 0; nt < 4; ++nt)
      acc[nt] = WMMA_BF16(aF[cur][0].bf, b[nt][0].bf, acc[nt]);
    #pragma unroll
    for (int nt = 0; nt < 4; ++nt)
      acc[nt] = WMMA_BF16(aF[cur][1].bf, b[nt][1].bf, acc[nt]);
  }
  (void)0;
}

// ---------------- Kernel 3: QKV GEMM ----------------------------------------
__global__ __launch_bounds__(256) void qkv_gemm_kernel(const u16* __restrict__ wbf,
                                                       const float* __restrict__ bias,
                                                       const u16* __restrict__ hT,
                                                       u16* __restrict__ qT,
                                                       u16* __restrict__ kT,
                                                       u16* __restrict__ vN) {
  __shared__ u16 Bs[2][64 * 72];
  const int lane = threadIdx.x & 31, w = threadIdx.x >> 5;
  const int l16 = lane & 15, half = lane >> 4;
  const int mw = blockIdx.x * 128 + w * 16;
  const int nb = blockIdx.y * 64;
  v8f acc[4] = {};
  gemm_core<int>(wbf + (mw + l16) * CCH, hT + nb * CCH, Bs, w, l16, half, acc);
  const int sec = mw >> 9;                 // 0=q 1=k 2=v (uniform per block)
  const int m0  = mw + half * 8;
  const int om  = m0 & (CCH - 1);
  const int head = om >> 6, d0 = om & 63;
  const float osc = (sec == 0) ? 0.125f : 1.0f;   // fold 64^-0.5 into Q
  float bv[8];
  #pragma unroll
  for (int r = 0; r < 8; ++r) bv[r] = bias[m0 + r];
  #pragma unroll
  for (int nt = 0; nt < 4; ++nt) {
    const int p = nb + nt * 16 + l16;
    if (sec < 2) {
      Pack8 pk;
      #pragma unroll
      for (int r = 0; r < 8; ++r) pk.u[r] = f2bf((acc[nt][r] + bv[r]) * osc);
      u16* dst = (sec == 0 ? qT : kT) + (head * HW + p) * HD + d0;
      *(uint4*)dst = pk.q;
    } else {
      #pragma unroll
      for (int r = 0; r < 8; ++r)
        vN[head * HD * HW + (d0 + r) * HW + p] = f2bf(acc[nt][r] + bv[r]);
    }
  }
}

// ---------------- Kernel 4: flash attention (TDM-staged K/V) ----------------
__global__ __launch_bounds__(256) void flash_attn_kernel(const u16* __restrict__ qT,
                                                         const u16* __restrict__ kT,
                                                         const u16* __restrict__ vN,
                                                         u16* __restrict__ aT) {
  __shared__ u16 Ks[2][64 * 72];           // 64 keys x 64 d (rows padded ->72)
  __shared__ u16 Vs[2][64 * 72];           // 64 d x 64 keys (rows padded ->72)
  __shared__ u16 Pl[8][16][72];            // per-wave P tile 16 x 64
  const int lane = threadIdx.x & 31, w = threadIdx.x >> 5;
  const int l16 = lane & 15, half = lane >> 4;
  const int head = blockIdx.y;
  const int q0 = blockIdx.x * 128 + w * 16;
  const u16* qrow = qT + ((head << 12) + q0 + l16) * HD;
  Frag aQ0, aQ1;
  aQ0.q[0] = *(const uint4*)(qrow + half * 8);
  aQ0.q[1] = *(const uint4*)(qrow + half * 8 + 16);
  aQ1.q[0] = *(const uint4*)(qrow + 32 + half * 8);
  aQ1.q[1] = *(const uint4*)(qrow + 32 + half * 8 + 16);
  v8f O[4] = {};
  float mi[8], li[8];
  #pragma unroll
  for (int r = 0; r < 8; ++r) { mi[r] = -3.0e38f; li[r] = 0.f; }
  const u16* kbase = kT + (head << 12) * HD;
  const u16* vbase = vN + head * HD * HW;
  if (w == 0) {
    tdm_load_2d(LDSOFF(&Ks[0][0]), kbase, HD, HW, HD, 64, 64);
    tdm_load_2d(LDSOFF(&Vs[0][0]), vbase, HW, HD, HW, 64, 64);
  }
  for (int ci = 0; ci < HW / 64; ++ci) {
    const int kc = ci * 64;
    const int cur = ci & 1;
    __syncthreads();
    if (w == 0) {
      if (ci + 1 < HW / 64) {
        tdm_load_2d(LDSOFF(&Ks[cur ^ 1][0]), kbase + (kc + 64) * HD,
                    HD, HW, HD, 64, 64);
        tdm_load_2d(LDSOFF(&Vs[cur ^ 1][0]), vbase + kc + 64,
                    HW, HD, HW, 64, 64);
        __builtin_amdgcn_s_wait_tensorcnt(2);
      } else {
        __builtin_amdgcn_s_wait_tensorcnt(0);
      }
    }
    __syncthreads();
    // S = Q K^T  (Q pre-scaled by 0.125 at QKV epilogue)
    Frag kb[4][2];
    #pragma unroll
    for (int nt = 0; nt < 4; ++nt) {
      const u16* kr = &Ks[cur][(nt * 16 + l16) * 72 + half * 8];
      kb[nt][0].q[0] = *(const uint4*)(kr);      kb[nt][0].q[1] = *(const uint4*)(kr + 16);
      kb[nt][1].q[0] = *(const uint4*)(kr + 32); kb[nt][1].q[1] = *(const uint4*)(kr + 48);
    }
    v8f s[4] = {};
    #pragma unroll
    for (int nt = 0; nt < 4; ++nt) s[nt] = WMMA_BF16(aQ0.bf, kb[nt][0].bf, s[nt]);
    #pragma unroll
    for (int nt = 0; nt < 4; ++nt) s[nt] = WMMA_BF16(aQ1.bf, kb[nt][1].bf, s[nt]);
    // online softmax: row r+8*half lives at element r across 16 lanes
    float al[8];
    #pragma unroll
    for (int r = 0; r < 8; ++r) {
      float mx = fmaxf(fmaxf(s[0][r], s[1][r]), fmaxf(s[2][r], s[3][r]));
      mx = fmaxf(mx, __shfl_xor(mx, 1));
      mx = fmaxf(mx, __shfl_xor(mx, 2));
      mx = fmaxf(mx, __shfl_xor(mx, 4));
      mx = fmaxf(mx, __shfl_xor(mx, 8));
      const float mn = fmaxf(mi[r], mx);
      const float alpha = __expf(mi[r] - mn);
      mi[r] = mn;
      const float p0 = __expf(s[0][r] - mn), p1 = __expf(s[1][r] - mn);
      const float p2 = __expf(s[2][r] - mn), p3 = __expf(s[3][r] - mn);
      s[0][r] = p0; s[1][r] = p1; s[2][r] = p2; s[3][r] = p3;
      float rs = (p0 + p1) + (p2 + p3);
      rs += __shfl_xor(rs, 1);
      rs += __shfl_xor(rs, 2);
      rs += __shfl_xor(rs, 4);
      rs += __shfl_xor(rs, 8);
      li[r] = li[r] * alpha + rs;
      al[r] = alpha;
    }
    #pragma unroll
    for (int nt = 0; nt < 4; ++nt)
      #pragma unroll
      for (int r = 0; r < 8; ++r) O[nt][r] *= al[r];
    // re-layout P (C-layout) -> A-fragment layout through wave-private LDS:
    // 32 ds_store_b16, then 16 independent V-fragment ds_load_b128, then a
    // partial wait (stores done, V loads still in flight), then P loads.
    #pragma unroll
    for (int nt = 0; nt < 4; ++nt)
      #pragma unroll
      for (int r = 0; r < 8; ++r)
        Pl[w][half * 8 + r][nt * 16 + l16] = f2bf(s[nt][r]);
    Frag vb[4][2];
    #pragma unroll
    for (int nt = 0; nt < 4; ++nt) {
      const u16* vr = &Vs[cur][(nt * 16 + l16) * 72 + half * 8];
      vb[nt][0].q[0] = *(const uint4*)(vr);      vb[nt][0].q[1] = *(const uint4*)(vr + 16);
      vb[nt][1].q[0] = *(const uint4*)(vr + 32); vb[nt][1].q[1] = *(const uint4*)(vr + 48);
    }
    asm volatile("s_wait_dscnt 0x10" ::: "memory");   // 32 stores retired
    Frag pa0, pa1;
    pa0.q[0] = *(const uint4*)(&Pl[w][l16][half * 8]);
    pa0.q[1] = *(const uint4*)(&Pl[w][l16][half * 8 + 16]);
    pa1.q[0] = *(const uint4*)(&Pl[w][l16][half * 8 + 32]);
    pa1.q[1] = *(const uint4*)(&Pl[w][l16][half * 8 + 48]);
    #pragma unroll
    for (int nt = 0; nt < 4; ++nt) O[nt] = WMMA_BF16(pa0.bf, vb[nt][0].bf, O[nt]);
    #pragma unroll
    for (int nt = 0; nt < 4; ++nt) O[nt] = WMMA_BF16(pa1.bf, vb[nt][1].bf, O[nt]);
    asm volatile("" ::: "memory");
  }
  float inv[8];
  #pragma unroll
  for (int r = 0; r < 8; ++r) inv[r] = 1.0f / li[r];
  #pragma unroll
  for (int nt = 0; nt < 4; ++nt) {
    const int c = head * HD + nt * 16 + l16;
    #pragma unroll
    for (int r = 0; r < 8; ++r)
      aT[(q0 + half * 8 + r) * CCH + c] = f2bf(O[nt][r] * inv[r]);
  }
}

// ---------------- Kernel 5: proj GEMM + bias + residual ---------------------
__global__ __launch_bounds__(256) void proj_gemm_kernel(const u16* __restrict__ wbf,
                                                        const float* __restrict__ bias,
                                                        const u16* __restrict__ aT,
                                                        const float* __restrict__ x,
                                                        float* __restrict__ out) {
  __shared__ u16 Bs[2][64 * 72];
  const int lane = threadIdx.x & 31, w = threadIdx.x >> 5;
  const int l16 = lane & 15, half = lane >> 4;
  const int mw = blockIdx.x * 128 + w * 16;
  const int nb = blockIdx.y * 64;
  v8f acc[4] = {};
  gemm_core<int>(wbf + (mw + l16) * CCH, aT + nb * CCH, Bs, w, l16, half, acc);
  const int m0 = mw + half * 8;
  float bv[8];
  #pragma unroll
  for (int r = 0; r < 8; ++r) bv[r] = bias[m0 + r];
  #pragma unroll
  for (int nt = 0; nt < 4; ++nt) {
    const int p = nb + nt * 16 + l16;
    #pragma unroll
    for (int r = 0; r < 8; ++r) {
      const int idx = (m0 + r) * HW + p;
      out[idx] = acc[nt][r] + bv[r] + x[idx];
    }
  }
}

// ---------------- launch ----------------------------------------------------
extern "C" void kernel_launch(void* const* d_in, const int* in_sizes, int n_in,
                              void* d_out, int out_size, void* d_ws, size_t ws_size,
                              hipStream_t stream) {
  const float* x      = (const float*)d_in[0];
  const float* norm_w = (const float*)d_in[1];
  const float* norm_b = (const float*)d_in[2];
  const float* qkv_w  = (const float*)d_in[3];
  const float* qkv_b  = (const float*)d_in[4];
  const float* proj_w = (const float*)d_in[5];
  const float* proj_b = (const float*)d_in[6];
  float* out = (float*)d_out;

  u16* ws   = (u16*)d_ws;
  u16* hT   = ws;                                  // 4096*512
  u16* qT   = hT + HW * CCH;                       // [head][seq][64] (pre-scaled)
  u16* kT   = qT + HW * CCH;
  u16* vN   = kT + HW * CCH;                       // [head][64][seq]
  u16* aT   = vN + HW * CCH;                       // 4096*512
  u16* qwbf = aT + HW * CCH;                       // 1536*512
  u16* pwbf = qwbf + 3 * CCH * CCH;                // 512*512

  gn_kernel<<<32, 256, 0, stream>>>(x, norm_w, norm_b, hT);
  cvt_w_kernel<<<(3 * CCH * CCH + CCH * CCH) / 256, 256, 0, stream>>>(qkv_w, proj_w, qwbf, pwbf);
  qkv_gemm_kernel<<<dim3(12, 64), 256, 0, stream>>>(qwbf, qkv_b, hT, qT, kT, vN);
  flash_attn_kernel<<<dim3(32, 8), 256, 0, stream>>>(qT, kT, vN, aT);
  proj_gemm_kernel<<<dim3(4, 64), 256, 0, stream>>>(pwbf, proj_b, aT, x, out);
}
